// Seq2Seq_55319178772629
// MI455X (gfx1250) — matrix operand order repaired and checked by reading, not compiled, not run on hardware
//
#include <hip/hip_runtime.h>
#include <hip/hip_bf16.h>
#include <stdint.h>

// ---------------------------------------------------------------------------
// Seq2Seq LSTM (enc 256 steps + dec 128 steps, B=256, H=1024, 2 layers)
// Persistent-kernel design for MI455X (gfx1250, wave32, WMMA):
//  * bf16 WMMA 16x16x32, f32 accumulate; weights pre-packed into B-fragment
//    order (~51MB bf16 -> fully L2-resident, HBM touched once)
//  * recurrent h state stored in bf16 (WMMA eats bf16 anyway -> zero extra
//    precision loss, and A-fragments become raw b128 loads, no cvt VALU)
//  * cell state c stays fp32 (it accumulates across 256 steps)
//  * grid-wide atomic barrier between layer phases (768 barriers total)
//  * LDS double-buffered staging of B fragments: 16 waves/WG share one read
//  * per-wave gate grouping (i,f,g,o of same hidden cols) -> in-register
//    LSTM cell update, no gate scratch
// ---------------------------------------------------------------------------

typedef __bf16 bf16_t;
typedef __attribute__((ext_vector_type(16))) __bf16 v16bf;
typedef __attribute__((ext_vector_type(8)))  __bf16 v8bf;
typedef __attribute__((ext_vector_type(8)))  float  v8f;
typedef __attribute__((ext_vector_type(4)))  float  v4f;

#define HID       1024
#define GATES     4096
#define BATCH     256
#define SRCT      256
#define PREDT     128
#define KC_L0     33      // 32 chunks of h (K=1024) + 1 zero-padded x chunk
#define KC_L1     64      // [h0_new (1024) | h1_prev (1024)]
#define NWG       32
#define WGTHREADS 512     // 16 waves = 16 batch M-tiles per WG

// fp32 -> bf16 round-to-nearest-even
static __device__ __forceinline__ bf16_t f2bf(float f) {
  unsigned u = __builtin_bit_cast(unsigned, f);
  u += 0x7fffu + ((u >> 16) & 1u);
  return __builtin_bit_cast(bf16_t, (unsigned short)(u >> 16));
}
static __device__ __forceinline__ float bf2f(bf16_t b) {
  unsigned u = ((unsigned)__builtin_bit_cast(unsigned short, b)) << 16;
  return __builtin_bit_cast(float, u);
}
static __device__ __forceinline__ float sigm(float x) {
  return 1.0f / (1.0f + __expf(-x));
}
static __device__ __forceinline__ float tanhfast(float x) {
  float e = __expf(-2.0f * x);
  return (1.0f - e) / (1.0f + e);
}

// ---------------------------------------------------------------------------
// Weight packing: B-fragment layout per ISA 7.12.2 (16-bit B 32x16):
//   col = lane%16, K = (i&7) + 16*(i>=8) + 8*(lane>=16), i = element 0..15 of v16bf.
// Combined K space: k < Ka -> Wa[col,k]; k-Ka < Kb -> Wb[col,k-Ka]; else 0.
// Blob layout: [colTile(256)][kChunk(KC)][lane(32)][16 bf16]
// ---------------------------------------------------------------------------
__global__ void pack_weights(const float* __restrict__ Wa, int Ka,
                             const float* __restrict__ Wb, int Kb,
                             int KC, bf16_t* __restrict__ out, long total) {
  long idx = (long)blockIdx.x * blockDim.x + threadIdx.x;
  if (idx >= total) return;
  int  i    = (int)(idx & 15);
  int  lane = (int)((idx >> 4) & 31);
  long rest = idx >> 9;
  int  kc   = (int)(rest % KC);
  int  ct   = (int)(rest / KC);
  int  col  = ct * 16 + (lane & 15);
  int  k    = kc * 32 + (i & 7) + ((i >= 8) ? 16 : 0) + ((lane >= 16) ? 8 : 0);
  float v = 0.0f;
  if (k < Ka)            v = Wa[(long)col * Ka + k];
  else if (k - Ka < Kb)  v = Wb[(long)col * Kb + (k - Ka)];
  out[idx] = f2bf(v);
}

__global__ void combine_bias(const float* __restrict__ a,
                             const float* __restrict__ b,
                             float* __restrict__ o) {
  int i = blockIdx.x * blockDim.x + threadIdx.x;
  if (i < GATES) o[i] = a[i] + b[i];
}

// A fragment from bf16 row-major h (ISA 7.12.2, 16-bit A 16x32):
// lane<16: elems 0..7 -> K=kb+0..7,  elems 8..15 -> K=kb+16..23
// lane>=16: +8 on K. Caller passes p already offset by hiK.
// Two 16B global loads, no conversion.
static __device__ __forceinline__ v16bf a_from_bf16(const bf16_t* __restrict__ p) {
  v8bf lo = *(const v8bf*)(p);
  v8bf hi = *(const v8bf*)(p + 16);
  return __builtin_shufflevector(lo, hi, 0,1,2,3,4,5,6,7,8,9,10,11,12,13,14,15);
}

// One layer-step for this WG: gates = A_cat @ W_packed + bias, then LSTM cell.
// WG owns hidden cols [nTile0*16, nTile0*16+32); wave wv owns batch rows
// [wv*16, wv*16+16). Each wave computes 8 C-tiles (2 hidden tiles x 4 gates).
template <int KC, bool HASB, int XIN>
static __device__ __forceinline__ void layer_phase(
    const bf16_t* __restrict__ pack,
    const bf16_t* __restrict__ hA,            // K chunks [0,32)
    const bf16_t* __restrict__ hB,            // K chunks [32,64) (HASB)
    const float* __restrict__ xr,             // x row (last chunk, !HASB)
    const float* __restrict__ bias,
    float* __restrict__ cbuf, bf16_t* __restrict__ hout,
    int nTile0, bf16_t* smem) {
  const int lane = threadIdx.x & 31;
  const int wv   = threadIdx.x >> 5;              // M tile (batch/16)
  const int arow = (wv << 4) + (lane & 15);       // batch row for A frag
  const int hiK  = (lane >= 16) ? 8 : 0;

  v8f acc[2][4];
#pragma unroll
  for (int nt = 0; nt < 2; ++nt)
#pragma unroll
    for (int g = 0; g < 4; ++g) {
      float bv = bias[g * HID + (nTile0 + nt) * 16 + (lane & 15)];
#pragma unroll
      for (int e = 0; e < 8; ++e) acc[nt][g][e] = bv;
    }

  // Cooperative LDS staging: 8 fragments/chunk (2 nt x 4 gates) = 8KB,
  // 512 threads x 16B. Double buffered.
  const int  bsel = threadIdx.x >> 6;             // 0..7 = nt*4+g
  const int  snt = bsel >> 2, sg = bsel & 3;
  const long sct = (long)(sg * 64 + nTile0 + snt);
  const uint4* gsrc = (const uint4*)(pack + sct * KC * 512) + (threadIdx.x & 63);

  uint4 stv = gsrc[0];
  int cur = 0;
  *((uint4*)(smem) + threadIdx.x) = stv;
  __syncthreads();

  for (int kc = 0; kc < KC; ++kc) {
    if (kc + 1 < KC) stv = gsrc[(long)(kc + 1) * 64];   // global prefetch

    // A fragment (global, overlaps with LDS loads below)
    v16bf a;
    if (HASB) {
      const bf16_t* hp = (kc < 32 ? hA : hB) +
                         (long)arow * HID + (long)(kc & 31) * 32 + hiK;
      a = a_from_bf16(hp);
    } else if (kc < 32) {
      a = a_from_bf16(hA + (long)arow * HID + (long)kc * 32 + hiK);
    } else {
      // zero-padded x chunk: only lanes<16, elems<XIN nonzero (XIN<=8)
      bf16_t z = __builtin_bit_cast(bf16_t, (unsigned short)0);
#pragma unroll
      for (int i = 0; i < 16; ++i) a[i] = z;
      if (lane < 16) {
#pragma unroll
        for (int i = 0; i < XIN; ++i) a[i] = f2bf(xr[i]);
      }
    }

    // load all 8 B fragments first, then MMA (partial dscnt waits)
    const bf16_t* sb = smem + cur * 4096;
    v16bf bm[8];
#pragma unroll
    for (int q = 0; q < 8; ++q)
      bm[q] = *(const v16bf*)(sb + (q * 32 + lane) * 16);
#pragma unroll
    for (int nt = 0; nt < 2; ++nt)
#pragma unroll
      for (int g = 0; g < 4; ++g)
        acc[nt][g] = __builtin_amdgcn_wmma_f32_16x16x32_bf16(
            false, a, false, bm[nt * 4 + g], (short)0, acc[nt][g], false, false);

    if (kc + 1 < KC)
      *((uint4*)(smem + (cur ^ 1) * 4096) + threadIdx.x) = stv;
    __syncthreads();
    cur ^= 1;
  }

  // In-register LSTM cell update. C layout: elem e -> row e+8*(lane>=16),
  // col lane%16. This wave exclusively owns these (row, col) cells.
#pragma unroll
  for (int nt = 0; nt < 2; ++nt) {
    int n = (nTile0 + nt) * 16 + (lane & 15);
#pragma unroll
    for (int e = 0; e < 8; ++e) {
      int  r   = (wv << 4) + e + ((lane >= 16) ? 8 : 0);
      long idx = (long)r * HID + n;
      float iv = sigm(acc[nt][0][e]);
      float fv = sigm(acc[nt][1][e]);
      float gv = tanhfast(acc[nt][2][e]);
      float ov = sigm(acc[nt][3][e]);
      float cn = fv * cbuf[idx] + iv * gv;
      cbuf[idx] = cn;
      hout[idx] = f2bf(ov * tanhfast(cn));
    }
  }
}

static __device__ __forceinline__ void grid_barrier(unsigned* cnt, unsigned* iter) {
  __threadfence();
  __syncthreads();
  if (threadIdx.x == 0) {
    unsigned target = gridDim.x * (++(*iter));
    __hip_atomic_fetch_add(cnt, 1u, __ATOMIC_RELEASE, __HIP_MEMORY_SCOPE_AGENT);
    while (__hip_atomic_load(cnt, __ATOMIC_ACQUIRE, __HIP_MEMORY_SCOPE_AGENT) < target)
      __builtin_amdgcn_s_sleep(2);
  }
  __syncthreads();
  __threadfence();
}

// out[b, t, d] = fc_b[d] + h1[b,:] . fc_W[d,:]   (512 threads of block 0)
static __device__ __forceinline__ void compute_fc(
    const bf16_t* __restrict__ h1buf, const float* __restrict__ fcW,
    const float* __restrict__ fcb, float* __restrict__ out, int t) {
  int b = threadIdx.x >> 1;
  int d = threadIdx.x & 1;
  const bf16_t* hr = h1buf + (long)b * HID;
  const float*  wr = fcW + (long)d * HID;
  float s = fcb[d];
  for (int k = 0; k < HID; k += 8) {
    v8bf hv = *(const v8bf*)(hr + k);
#pragma unroll
    for (int j = 0; j < 8; ++j) s += bf2f(hv[j]) * wr[k + j];
  }
  out[((long)b * PREDT + t) * 2 + d] = s;
}

template <int DEC, int XIN>
__global__ __launch_bounds__(WGTHREADS, 1)
void lstm_seq(const bf16_t* __restrict__ packL0, const bf16_t* __restrict__ packL1,
              const float* __restrict__ bias0, const float* __restrict__ bias1,
              bf16_t* __restrict__ h0, bf16_t* __restrict__ h1,
              float* __restrict__ c0, float* __restrict__ c1,
              const float* __restrict__ src, const float* __restrict__ tgt,
              int steps,
              const float* __restrict__ fcW, const float* __restrict__ fcb,
              float* __restrict__ out, unsigned* cnt) {
  __shared__ __align__(32) bf16_t smem[2 * 4096];   // 16KB double buffer
  const long STATE = (long)BATCH * HID;
  const int  lane   = threadIdx.x & 31;
  const int  wv     = threadIdx.x >> 5;
  const int  nTile0 = 2 * blockIdx.x;               // hidden tiles {2w, 2w+1}
  const int  brow   = (wv << 4) + (lane & 15);
  unsigned biter = 0;

  for (int t = 0; t < steps; ++t) {
    const bf16_t* hp0 = h0 + (t & 1) * STATE;        // read (prev)
    bf16_t*       hn0 = h0 + ((t + 1) & 1) * STATE;  // write (new)
    const bf16_t* hp1 = h1 + (t & 1) * STATE;
    bf16_t*       hn1 = h1 + ((t + 1) & 1) * STATE;

    // overlap: decoder FC for step t-1 while other WGs run layer 0
    if (DEC && t > 0 && blockIdx.x == 0)
      compute_fc(hp1, fcW, fcb, out, t - 1);

    // layer 0 input row for this lane (teacher forcing for decoder)
    const float* xr;
    if (!DEC)          xr = src + ((long)brow * SRCT + t) * 8;
    else if (t == 0)   xr = src + ((long)brow * SRCT + (SRCT - 1)) * 8;
    else               xr = tgt + ((long)brow * PREDT + (t - 1)) * 2;

    // layer 0: gates = [h0_prev | x_t] @ [Whh0 ; Wih0]^T + b
    layer_phase<KC_L0, false, XIN>(packL0, hp0, nullptr, xr, bias0,
                                   c0, hn0, nTile0, smem);
    grid_barrier(cnt, &biter);

    // layer 1: gates = [h0_new | h1_prev] @ [Wih1 ; Whh1]^T + b
    layer_phase<KC_L1, true, 0>(packL1, hn0, hp1, nullptr, bias1,
                                c1, hn1, nTile0, smem);
    grid_barrier(cnt, &biter);
  }
  if (DEC && blockIdx.x == 0)
    compute_fc(h1 + (steps & 1) * STATE, fcW, fcb, out, steps - 1);
}

// ---------------------------------------------------------------------------
extern "C" void kernel_launch(void* const* d_in, const int* in_sizes, int n_in,
                              void* d_out, int out_size, void* d_ws, size_t ws_size,
                              hipStream_t stream) {
  (void)in_sizes; (void)n_in; (void)out_size; (void)ws_size;
  const float* src   = (const float*)d_in[0];
  const float* tgt   = (const float*)d_in[1];
  const float* eWih0 = (const float*)d_in[2];
  const float* eWhh0 = (const float*)d_in[3];
  const float* ebih0 = (const float*)d_in[4];
  const float* ebhh0 = (const float*)d_in[5];
  const float* eWih1 = (const float*)d_in[6];
  const float* eWhh1 = (const float*)d_in[7];
  const float* ebih1 = (const float*)d_in[8];
  const float* ebhh1 = (const float*)d_in[9];
  const float* dWih0 = (const float*)d_in[10];
  const float* dWhh0 = (const float*)d_in[11];
  const float* dbih0 = (const float*)d_in[12];
  const float* dbhh0 = (const float*)d_in[13];
  const float* dWih1 = (const float*)d_in[14];
  const float* dWhh1 = (const float*)d_in[15];
  const float* dbih1 = (const float*)d_in[16];
  const float* dbhh1 = (const float*)d_in[17];
  const float* fcW   = (const float*)d_in[18];
  const float* fcb   = (const float*)d_in[19];

  uint8_t* w = (uint8_t*)d_ws;
  size_t off = 0;
  auto take = [&](size_t bytes) -> void* {
    void* p = w + off;
    off = (off + bytes + 255) & ~(size_t)255;
    return p;
  };

  const long STATE = (long)BATCH * HID;
  unsigned* cnt = (unsigned*)take(256);
  bf16_t* h0 = (bf16_t*)take(2 * STATE * 2);   // bf16 double-buffered h
  bf16_t* h1 = (bf16_t*)take(2 * STATE * 2);
  float*  c0 = (float*)take(STATE * 4);        // fp32 cell state
  float*  c1 = (float*)take(STATE * 4);
  float* bE0 = (float*)take(GATES * 4);
  float* bE1 = (float*)take(GATES * 4);
  float* bD0 = (float*)take(GATES * 4);
  float* bD1 = (float*)take(GATES * 4);
  const long P0 = 256L * KC_L0 * 512;   // bf16 elems per layer-0 pack
  const long P1 = 256L * KC_L1 * 512;
  bf16_t* pE0 = (bf16_t*)take((size_t)P0 * 2);
  bf16_t* pE1 = (bf16_t*)take((size_t)P1 * 2);
  bf16_t* pD0 = (bf16_t*)take((size_t)P0 * 2);
  bf16_t* pD1 = (bf16_t*)take((size_t)P1 * 2);

  combine_bias<<<16, 256, 0, stream>>>(ebih0, ebhh0, bE0);
  combine_bias<<<16, 256, 0, stream>>>(ebih1, ebhh1, bE1);
  combine_bias<<<16, 256, 0, stream>>>(dbih0, dbhh0, bD0);
  combine_bias<<<16, 256, 0, stream>>>(dbih1, dbhh1, bD1);

  pack_weights<<<(int)((P0 + 255) / 256), 256, 0, stream>>>(eWhh0, 1024, eWih0, 8,    KC_L0, pE0, P0);
  pack_weights<<<(int)((P1 + 255) / 256), 256, 0, stream>>>(eWih1, 1024, eWhh1, 1024, KC_L1, pE1, P1);
  pack_weights<<<(int)((P0 + 255) / 256), 256, 0, stream>>>(dWhh0, 1024, dWih0, 2,    KC_L0, pD0, P0);
  pack_weights<<<(int)((P1 + 255) / 256), 256, 0, stream>>>(dWih1, 1024, dWhh1, 1024, KC_L1, pD1, P1);

  hipMemsetAsync(h0, 0, 2 * STATE * 2, stream);
  hipMemsetAsync(h1, 0, 2 * STATE * 2, stream);
  hipMemsetAsync(c0, 0, STATE * 4, stream);
  hipMemsetAsync(c1, 0, STATE * 4, stream);
  hipMemsetAsync(cnt, 0, 4, stream);

  // encoder (no output head)
  lstm_seq<0, 8><<<NWG, WGTHREADS, 0, stream>>>(pE0, pE1, bE0, bE1, h0, h1, c0, c1,
                                                src, nullptr, SRCT,
                                                nullptr, nullptr, nullptr, cnt);
  hipMemsetAsync(cnt, 0, 4, stream);
  // decoder (continues from encoder h/c, teacher-forced inputs, FC head)
  lstm_seq<1, 2><<<NWG, WGTHREADS, 0, stream>>>(pD0, pD1, bD0, bD1, h0, h1, c0, c1,
                                                src, tgt, PREDT,
                                                fcW, fcb, (float*)d_out, cnt);
}
